// DFHGNN_59708635349494
// MI455X (gfx1250) — matrix operand order repaired and checked by reading, not compiled, not run on hardware
//
#include <hip/hip_runtime.h>

// ---------------------------------------------------------------------------
// DF-HGNN forward for MI455X (gfx1250).
// Heavy GEMMs -> v_wmma_f32_16x16x32_bf16 (wave32), H converted to bf16 once.
// All B operands kept transposed (Bt[N][K]) so LDS staging is pure b128;
// LDS double-buffered (1 barrier / K-step); split-K for the Ht@Y GEMM.
// Fragment loads hoisted ahead of the WMMA chain for back-to-back issue.
// ---------------------------------------------------------------------------

#define N_NODES 20000
#define N_EDGES 4096
#define IN_DIM  512
#define DET_DIM 256
#define HID     256
#define HALF    128
#define OUT_DIM 2
#define EPSV    1e-6f
#define KSPLIT  5            // He GEMM K chunks: 20000 = 5 * 4000, 4000 % 32 == 0

typedef __attribute__((ext_vector_type(16))) __bf16         v16bf;
typedef __attribute__((ext_vector_type(8)))  float          v8f;
typedef __attribute__((ext_vector_type(8)))  unsigned short us8;
typedef __attribute__((ext_vector_type(4)))  unsigned short us4;
typedef __attribute__((ext_vector_type(4)))  float          f4;

union Frag { v16bf v; us8 h[2]; };

__device__ __forceinline__ unsigned short f32_to_bf16(float f) {
  unsigned int u = __float_as_uint(f);
  u += 0x7FFFu + ((u >> 16) & 1u);          // round-to-nearest-even
  return (unsigned short)(u >> 16);
}
__device__ __forceinline__ float bf16_to_f32(unsigned short h) {
  return __uint_as_float(((unsigned int)h) << 16);
}

// ---------------------------------------------------------------------------
// Tiled bf16 WMMA GEMM:  C = act( (A @ B) * rowscale[m] + bias[n] )
//   A  bf16 row-major [M,K]   (TRANSA: A[m,k] = Aptr[k*lda + m])
//   BT bf16 row-major [N,K]   (i.e. B stored transposed)
// Block tile: BM=32, BN=WN*64, BK=32; 256 threads = 8 waves (2 x 4);
// each wave: one A-frag reused over WN WMMAs (16 x WN*16 strip).
// OUTMODE: 0 = f32 [m][n] (+ blockIdx.z*M*ldc partial offset for split-K)
//          1 = bf16 [m][n]
//          2 = bf16 transposed [n][m]  (one b128 store per tile row-group)
// LDS stride 40 ushorts (80B): 16B-aligned rows, near-conflict-free b128.
// ---------------------------------------------------------------------------
template<int WN, bool TRANSA, bool HAS_BIAS, bool HAS_RS, bool RELU, int OUTMODE>
__global__ void __launch_bounds__(256)
gemm_k(const unsigned short* __restrict__ A, const unsigned short* __restrict__ BT,
       void* __restrict__ Cv, const float* __restrict__ bias,
       const float* __restrict__ rowscale,
       int K, int lda, int ldbT, int ldc, int M)
{
  constexpr int BN = WN * 64;
  constexpr int LS = 40;
  __shared__ __align__(16) unsigned short As[2][32][LS];
  __shared__ __align__(16) unsigned short Bs[2][BN][LS];

  const int tid  = threadIdx.x;
  const int lane = tid & 31;
  const int wave = tid >> 5;
  const int wm = wave >> 2, wn = wave & 3;
  const int blockM = blockIdx.y * 32;
  const int blockN = blockIdx.x * BN;
  const int kBase  = blockIdx.z * K;          // split-K chunk base

  // staging thread mappings
  int a_m, a_k;
  if (TRANSA) { a_k = tid >> 3; a_m = (tid & 7) * 4; }   // contiguous m in global
  else        { a_m = tid >> 3; a_k = (tid & 7) * 4; }   // contiguous k in global
  const int bn_ = tid >> 2;          // 0..63: BT row within pass
  const int bk_ = (tid & 3) * 8;     // 0,8,16,24: k offset (us8)

  us4 aReg;
  us8 bReg[WN];

  auto loadT = [&](int k0) {
    if (TRANSA) aReg = *(const us4*)(A + (size_t)(kBase + k0 + a_k) * lda + (blockM + a_m));
    else        aReg = *(const us4*)(A + (size_t)(blockM + a_m) * lda + (kBase + k0 + a_k));
    #pragma unroll
    for (int p = 0; p < WN; ++p)
      bReg[p] = *(const us8*)(BT + (size_t)(blockN + bn_ + p * 64) * ldbT
                                 + (kBase + k0 + bk_));
  };
  auto storeT = [&](int buf) {
    if (TRANSA) {
      #pragma unroll
      for (int j = 0; j < 4; ++j) As[buf][a_m + j][a_k] = aReg[j];
    } else {
      *(us4*)&As[buf][a_m][a_k] = aReg;
    }
    #pragma unroll
    for (int p = 0; p < WN; ++p)
      *(us8*)&Bs[buf][bn_ + p * 64][bk_] = bReg[p];    // already transposed
  };

  v8f acc[WN] = {};
  const int T = K / 32;

  loadT(0);
  storeT(0);
  __syncthreads();
  if (T > 1) loadT(32);

  // fragment addresses (contiguous k order: k = 16*(lane/16) + s, A and B alike)
  const int arow  = wm * 16 + (lane & 15);
  const int akb   = (lane < 16) ? 0 : 16;
  const int bkb   = (lane < 16) ? 0 : 16;
  const int bcol0 = wn * WN * 16 + (lane & 15);

  for (int i = 0; i < T; ++i) {
    const int cur = i & 1;
    // hoist ALL fragment loads ahead of the WMMA chain: the DS pipe issues
    // 2*(WN+1) b128 loads as one batch, WMMAs then run back-to-back.
    Frag a, b[WN];
    a.h[0] = *(const us8*)&As[cur][arow][akb];         // ds_load_b128
    a.h[1] = *(const us8*)&As[cur][arow][akb + 8];
    #pragma unroll
    for (int t = 0; t < WN; ++t) {
      b[t].h[0] = *(const us8*)&Bs[cur][bcol0 + t * 16][bkb];
      b[t].h[1] = *(const us8*)&Bs[cur][bcol0 + t * 16][bkb + 8];
    }
    #pragma unroll
    for (int t = 0; t < WN; ++t)
      acc[t] = __builtin_amdgcn_wmma_f32_16x16x32_bf16(
          false, a.v, false, b[t].v, (short)0, acc[t], false, false);
    if (i + 1 < T) {
      storeT(1 - cur);                  // regs hold tile i+1; safe: last reads
      __syncthreads();                  //   of that buffer fenced at iter i-1
      if (i + 2 < T) loadT((i + 2) * 32);
    }
  }

  // epilogue: C VGPR r -> (lanes 0-15) M=r, (lanes 16-31) M=8+r; N=lane&15
  const int m_base = blockM + wm * 16 + ((lane < 16) ? 0 : 8);
  float rs[8];
  if (HAS_RS) {
    #pragma unroll
    for (int r = 0; r < 8; ++r) rs[r] = rowscale[m_base + r];
  }
  #pragma unroll
  for (int t = 0; t < WN; ++t) {
    const int n = blockN + wn * WN * 16 + t * 16 + (lane & 15);
    const float bv = HAS_BIAS ? bias[n] : 0.0f;
    if (OUTMODE == 2) {
      us8 o;
      #pragma unroll
      for (int r = 0; r < 8; ++r) {
        float v = acc[t][r];
        if (HAS_RS)   v *= rs[r];
        if (HAS_BIAS) v += bv;
        if (RELU)     v = fmaxf(v, 0.0f);
        o[r] = f32_to_bf16(v);
      }
      *(us8*)((unsigned short*)Cv + (size_t)n * ldc + m_base) = o;  // m-contig
    } else {
      #pragma unroll
      for (int r = 0; r < 8; ++r) {
        float v = acc[t][r];
        if (HAS_RS)   v *= rs[r];
        if (HAS_BIAS) v += bv;
        if (RELU)     v = fmaxf(v, 0.0f);
        if (OUTMODE == 0)
          ((float*)Cv)[(size_t)blockIdx.z * M * ldc + (size_t)(m_base + r) * ldc + n] = v;
        else
          ((unsigned short*)Cv)[(size_t)(m_base + r) * ldc + n] = f32_to_bf16(v);
      }
    }
  }
}

// ---------------------------------------------------------------------------
// Elementwise / reduction helpers
// ---------------------------------------------------------------------------
__global__ void cvt4_k(const f4* __restrict__ src, us4* __restrict__ dst, long n4)
{
  long i = (long)blockIdx.x * blockDim.x + threadIdx.x;
  const long stride = (long)gridDim.x * blockDim.x;
  for (; i < n4; i += stride) {
    f4 v = src[i];
    us4 o;
    #pragma unroll
    for (int j = 0; j < 4; ++j) o[j] = f32_to_bf16(v[j]);
    dst[i] = o;
  }
}

// wT[n][k] = bf16(w[k][n]) — small weight matrices only
__global__ void cvtT_k(const float* __restrict__ w, unsigned short* __restrict__ wT,
                       int K, int N)
{
  const int idx = blockIdx.x * 256 + threadIdx.x;
  if (idx >= K * N) return;
  const int n = idx / K, k = idx % K;
  wT[idx] = f32_to_bf16(w[(size_t)k * N + n]);
}

// De partial column-sums: grid (E/256, 40), each block sums 500 rows
__global__ void edge_partial_k(const float* __restrict__ H, float* __restrict__ Dep)
{
  const int e   = blockIdx.x * 256 + threadIdx.x;
  const int seg = blockIdx.y;
  const float* Hp = H + (size_t)seg * 500 * N_EDGES + e;
  float s = 0.f;
  for (int r = 0; r < 500; ++r) s += Hp[(size_t)r * N_EDGES];
  Dep[seg * N_EDGES + e] = s;
}
__global__ void edge_final_k(const float* __restrict__ Dep,
                             const float* __restrict__ ew, float* __restrict__ we)
{
  const int e = blockIdx.x * 256 + threadIdx.x;
  float s = 0.f;
  for (int i = 0; i < 40; ++i) s += Dep[i * N_EDGES + e];
  we[e] = ew[e] / fmaxf(s, EPSV);
}

// isd[n] = rsqrt(max(H[n,:] . w, eps)); one wave32 per row, f4 loads
__global__ void __launch_bounds__(256)
node_scale_k(const float* __restrict__ H, const float* __restrict__ ew,
             float* __restrict__ isd)
{
  const int row  = blockIdx.x * 8 + (threadIdx.x >> 5);
  const int lane = threadIdx.x & 31;
  const f4* Hr = (const f4*)(H + (size_t)row * N_EDGES);
  const f4* w4 = (const f4*)ew;
  float s = 0.f;
  for (int i = lane; i < N_EDGES / 4; i += 32) {
    f4 h = Hr[i], w = w4[i];
    s += h[0]*w[0] + h[1]*w[1] + h[2]*w[2] + h[3]*w[3];
  }
  for (int off = 16; off > 0; off >>= 1) s += __shfl_down(s, off, 32);
  if (lane == 0) isd[row] = rsqrtf(fmaxf(s, EPSV));
}

// HebT[n][e] = bf16( we[e] * sum_z Hep[z][e][n] )  — deterministic K-split reduce
__global__ void he_reduce_k(const float* __restrict__ Hep,
                            const float* __restrict__ we,
                            unsigned short* __restrict__ HebT)
{
  const int idx = blockIdx.x * 256 + threadIdx.x;   // over HID*N_EDGES = 1M
  const int n = idx >> 12;          // / N_EDGES
  const int e = idx & (N_EDGES - 1);
  float s = 0.f;
  #pragma unroll
  for (int z = 0; z < KSPLIT; ++z)
    s += Hep[(size_t)z * N_EDGES * HID + (size_t)e * HID + n];
  HebT[(size_t)n * N_EDGES + e] = f32_to_bf16(s * we[e]);
}

// cat = [proj_x | proj_z] as bf16
__global__ void pack_cat_k(const float* __restrict__ px, const float* __restrict__ pz,
                           unsigned short* __restrict__ cat, long total)
{
  long i = (long)blockIdx.x * blockDim.x + threadIdx.x;
  const long stride = (long)gridDim.x * blockDim.x;
  for (; i < total; i += stride) {
    const long row = i >> 7; const int col = (int)(i & 127);
    cat[row * (2 * HALF) + col]        = f32_to_bf16(px[i]);
    cat[row * (2 * HALF) + HALF + col] = f32_to_bf16(pz[i]);
  }
}

// gate = sigmoid(t2) -> f32 out; fused = gate*pz + (1-gate)*px -> bf16
__global__ void gate_fuse_k(const float* __restrict__ t2,
                            const float* __restrict__ px, const float* __restrict__ pz,
                            float* __restrict__ gate_out,
                            unsigned short* __restrict__ fused, long total)
{
  long i = (long)blockIdx.x * blockDim.x + threadIdx.x;
  const long stride = (long)gridDim.x * blockDim.x;
  for (; i < total; i += stride) {
    const float g = 1.0f / (1.0f + expf(-t2[i]));
    gate_out[i] = g;
    fused[i] = f32_to_bf16(g * pz[i] + (1.0f - g) * px[i]);
  }
}

// logits = h2 @ out_w + out_b  (OUT=2); one wave32 per row
__global__ void __launch_bounds__(256)
logits_k(const unsigned short* __restrict__ h2, const float* __restrict__ ow,
         const float* __restrict__ ob, float* __restrict__ out)
{
  const int row  = blockIdx.x * 8 + (threadIdx.x >> 5);
  const int lane = threadIdx.x & 31;
  float s0 = 0.f, s1 = 0.f;
  for (int k = lane; k < HID; k += 32) {
    const float h = bf16_to_f32(h2[(size_t)row * HID + k]);
    s0 += h * ow[k * OUT_DIM + 0];
    s1 += h * ow[k * OUT_DIM + 1];
  }
  for (int off = 16; off > 0; off >>= 1) {
    s0 += __shfl_down(s0, off, 32);
    s1 += __shfl_down(s1, off, 32);
  }
  if (lane == 0) {
    out[(size_t)row * OUT_DIM + 0] = s0 + ob[0];
    out[(size_t)row * OUT_DIM + 1] = s1 + ob[1];
  }
}

// ---------------------------------------------------------------------------
// Host orchestration
// ---------------------------------------------------------------------------
extern "C" void kernel_launch(void* const* d_in, const int* in_sizes, int n_in,
                              void* d_out, int out_size, void* d_ws, size_t ws_size,
                              hipStream_t stream)
{
  (void)in_sizes; (void)n_in; (void)out_size; (void)ws_size;

  const float* x     = (const float*)d_in[0];
  const float* z     = (const float*)d_in[1];
  const float* H     = (const float*)d_in[2];
  const float* ew    = (const float*)d_in[3];
  const float* psi_w = (const float*)d_in[4];
  const float* psi_b = (const float*)d_in[5];
  const float* phi_w = (const float*)d_in[6];
  const float* phi_b = (const float*)d_in[7];
  const float* g1_w  = (const float*)d_in[8];
  const float* g1_b  = (const float*)d_in[9];
  const float* g2_w  = (const float*)d_in[10];
  const float* g2_b  = (const float*)d_in[11];
  const float* th1   = (const float*)d_in[12];
  const float* b1    = (const float*)d_in[13];
  const float* th2   = (const float*)d_in[14];
  const float* b2    = (const float*)d_in[15];
  const float* out_w = (const float*)d_in[16];
  const float* out_b = (const float*)d_in[17];

  float* logits = (float*)d_out;                               // [N,2]
  float* gate   = (float*)d_out + (size_t)N_NODES * OUT_DIM;   // [N,128]

  // workspace carve-out
  char*  ws  = (char*)d_ws;
  size_t off = 0;
  auto alloc = [&](size_t bytes) -> void* {
    void* p = ws + off; off += (bytes + 255) & ~(size_t)255; return p;
  };
  unsigned short* Hb     = (unsigned short*)alloc((size_t)N_NODES * N_EDGES * 2);
  unsigned short* xb     = (unsigned short*)alloc((size_t)N_NODES * IN_DIM * 2);
  unsigned short* zb     = (unsigned short*)alloc((size_t)N_NODES * DET_DIM * 2);
  unsigned short* psiwT  = (unsigned short*)alloc((size_t)IN_DIM * HALF * 2);
  unsigned short* phiwT  = (unsigned short*)alloc((size_t)DET_DIM * HALF * 2);
  unsigned short* g1wT   = (unsigned short*)alloc((size_t)2 * HALF * HID * 2);
  unsigned short* g2wT   = (unsigned short*)alloc((size_t)HID * HALF * 2);
  unsigned short* th1T   = (unsigned short*)alloc((size_t)HALF * HID * 2);
  unsigned short* th2T   = (unsigned short*)alloc((size_t)HID * HID * 2);
  float*          proj_x = (float*)alloc((size_t)N_NODES * HALF * 4);
  float*          proj_z = (float*)alloc((size_t)N_NODES * HALF * 4);
  unsigned short* catb   = (unsigned short*)alloc((size_t)N_NODES * 2 * HALF * 2);
  unsigned short* t1b    = (unsigned short*)alloc((size_t)N_NODES * HID * 2);
  float*          t2f    = (float*)alloc((size_t)N_NODES * HALF * 4);
  unsigned short* fusedb = (unsigned short*)alloc((size_t)N_NODES * HALF * 2);
  float*          Dep    = (float*)alloc((size_t)40 * N_EDGES * 4);
  float*          we     = (float*)alloc((size_t)N_EDGES * 4);
  float*          isd    = (float*)alloc((size_t)N_NODES * 4);
  unsigned short* YbT    = (unsigned short*)alloc((size_t)HID * N_NODES * 2);
  float*          Hep    = (float*)alloc((size_t)KSPLIT * N_EDGES * HID * 4);
  unsigned short* HebT   = (unsigned short*)alloc((size_t)HID * N_EDGES * 2);
  unsigned short* h1b    = (unsigned short*)alloc((size_t)N_NODES * HID * 2);
  unsigned short* h2b    = (unsigned short*)alloc((size_t)N_NODES * HID * 2);

  auto cvt = [&](const float* src, unsigned short* dst, long n) {
    long n4 = n / 4;
    long blocks = (n4 + 255) / 256; if (blocks > 4096) blocks = 4096;
    cvt4_k<<<dim3((unsigned)blocks), dim3(256), 0, stream>>>((const f4*)src, (us4*)dst, n4);
  };
  auto cvtT = [&](const float* w, unsigned short* wT, int K, int N) {
    cvtT_k<<<dim3((K * N + 255) / 256), dim3(256), 0, stream>>>(w, wT, K, N);
  };

  // bf16 conversions (H once; reused by the 4 big GEMMs); weights transposed
  cvt(H, Hb, (long)N_NODES * N_EDGES);
  cvt(x, xb, (long)N_NODES * IN_DIM);
  cvt(z, zb, (long)N_NODES * DET_DIM);
  cvtT(psi_w, psiwT, IN_DIM, HALF);
  cvtT(phi_w, phiwT, DET_DIM, HALF);
  cvtT(g1_w, g1wT, 2 * HALF, HID);
  cvtT(g2_w, g2wT, HID, HALF);
  cvtT(th1, th1T, HALF, HID);
  cvtT(th2, th2T, HID, HID);

  // degree scalings
  edge_partial_k<<<dim3(N_EDGES / 256, 40), dim3(256), 0, stream>>>(H, Dep);
  edge_final_k<<<dim3(N_EDGES / 256), dim3(256), 0, stream>>>(Dep, ew, we);
  node_scale_k<<<dim3(N_NODES / 8), dim3(256), 0, stream>>>(H, ew, isd);

  const dim3 gN(1, N_NODES / 32, 1);             // 625 M-blocks
  const dim3 gHe(2, N_EDGES / 32, KSPLIT);       // 2 x 128 x 5 = 1280 blocks
  const dim3 blk(256);

  // projections + gate MLP
  gemm_k<2,false,true,false,false,0><<<gN, blk, 0, stream>>>(
      xb, psiwT, proj_x, psi_b, nullptr, IN_DIM, IN_DIM, IN_DIM, HALF, N_NODES);
  gemm_k<2,false,true,false,false,0><<<gN, blk, 0, stream>>>(
      zb, phiwT, proj_z, phi_b, nullptr, DET_DIM, DET_DIM, DET_DIM, HALF, N_NODES);
  pack_cat_k<<<dim3(4096), blk, 0, stream>>>(proj_x, proj_z, catb, (long)N_NODES * HALF);
  gemm_k<4,false,true,false,true,1><<<gN, blk, 0, stream>>>(
      catb, g1wT, t1b, g1_b, nullptr, 2 * HALF, 2 * HALF, 2 * HALF, HID, N_NODES);
  gemm_k<2,false,true,false,false,0><<<gN, blk, 0, stream>>>(
      t1b, g2wT, t2f, g2_b, nullptr, HID, HID, HID, HALF, N_NODES);
  gate_fuse_k<<<dim3(4096), blk, 0, stream>>>(t2f, proj_x, proj_z, gate, fusedb,
                                              (long)N_NODES * HALF);

  // hconv layer 1: relu( (H @ (we*(Ht @ (isd*(fused@th1))))) * isd + b1 )
  gemm_k<4,false,false,true,false,2><<<gN, blk, 0, stream>>>(      // Yt = ((X@th1)*isd)^T
      fusedb, th1T, YbT, nullptr, isd, HALF, HALF, HALF, N_NODES, N_NODES);
  gemm_k<2,true,false,false,false,0><<<gHe, blk, 0, stream>>>(     // partial Ht@Y
      Hb, YbT, Hep, nullptr, nullptr, N_NODES / KSPLIT, N_EDGES, N_NODES, HID, N_EDGES);
  he_reduce_k<<<dim3(HID * N_EDGES / 256), blk, 0, stream>>>(Hep, we, HebT);
  gemm_k<4,false,true,true,true,1><<<gN, blk, 0, stream>>>(        // relu((H@He)*isd+b1)
      Hb, HebT, h1b, b1, isd, N_EDGES, N_EDGES, N_EDGES, HID, N_NODES);

  // hconv layer 2
  gemm_k<4,false,false,true,false,2><<<gN, blk, 0, stream>>>(
      h1b, th2T, YbT, nullptr, isd, HID, HID, HID, N_NODES, N_NODES);
  gemm_k<2,true,false,false,false,0><<<gHe, blk, 0, stream>>>(
      Hb, YbT, Hep, nullptr, nullptr, N_NODES / KSPLIT, N_EDGES, N_NODES, HID, N_EDGES);
  he_reduce_k<<<dim3(HID * N_EDGES / 256), blk, 0, stream>>>(Hep, we, HebT);
  gemm_k<4,false,true,true,true,1><<<gN, blk, 0, stream>>>(
      Hb, HebT, h2b, b2, isd, N_EDGES, N_EDGES, N_EDGES, HID, N_NODES);

  // output head
  logits_k<<<dim3(N_NODES / 8), blk, 0, stream>>>(h2b, out_w, out_b, logits);
}
